// SMoRESwitchGate_20057497272796
// MI455X (gfx1250) — matrix operand order repaired and checked by reading, not compiled, not run on hardware
//
#include <hip/hip_runtime.h>

typedef __attribute__((ext_vector_type(16))) __bf16 v16bf;
typedef __attribute__((ext_vector_type(8)))  __bf16 v8bf;
typedef __attribute__((ext_vector_type(8)))  float  v8f;
typedef __attribute__((ext_vector_type(4)))  float  v4f;
typedef __attribute__((ext_vector_type(4)))  int    v4i;

#define DIM_IN      4096
#define DIM_HID     1024
#define DIM_OUT     1024
#define NUM_EXPERTS 64
#define NUM_ACTIVE  8
#define NTOK        16384   // B*S = 4*4096
#define BALANCE_W   0.01f

#if __has_builtin(__builtin_amdgcn_global_load_async_to_lds_b128)
#define HAS_ASYNC_LDS 1
#else
#define HAS_ASYNC_LDS 0
#endif

__device__ __forceinline__ void wait_async_lds() {
#if __has_builtin(__builtin_amdgcn_s_wait_asynccnt)
  __builtin_amdgcn_s_wait_asynccnt(0);
#else
  asm volatile("s_wait_asynccnt 0" ::: "memory");
#endif
}

// ---------------------------------------------------------------------------
// fp32 -> bf16 conversion (weights only; x is converted in-register in GEMM1)
// ---------------------------------------------------------------------------
__global__ void f32_to_bf16_kernel(const float* __restrict__ src,
                                   __bf16* __restrict__ dst, int n) {
  int i = blockIdx.x * blockDim.x + threadIdx.x;
  int stride = gridDim.x * blockDim.x;
  for (; i < n; i += stride) dst[i] = (__bf16)src[i];
}

__global__ void zero_accum_kernel(float* __restrict__ acc) {
  if (threadIdx.x < 129) acc[threadIdx.x] = 0.f;
}

// ---------------------------------------------------------------------------
// bf16 WMMA GEMM:  C[M,Nout] = act(A[M,K] * W[Nout,K]^T + bias)
// block = 256 threads (8 waves), tile = 256 rows x 64 cols
// wave w computes rows [blk*256 + w*32, +32) x 64 cols: 2 M-tiles x 4 N-tiles
//   -> 8 WMMAs per 32-wide K step, B fragments reused across both M-tiles.
// W k-tile (64 x 32 bf16) double-buffered in LDS; staged with async
// global->LDS copies (ASYNCcnt) when available.
// ---------------------------------------------------------------------------
template <bool A_BF16, bool OUT_BF16, bool RELU>
__global__ __launch_bounds__(256) void gemm_wmma_kernel(
    const void* __restrict__ Av, const __bf16* __restrict__ W,
    const float* __restrict__ bias, void* __restrict__ Cv,
    int M, int K, int Nout) {
  constexpr int ROWP = 40;                 // LDS row pitch (elems): 80B, 16B-aligned, bank-spread
  __shared__ __bf16 lw[2][64 * ROWP];

  const int lane    = threadIdx.x & 31;
  const int wave    = threadIdx.x >> 5;
  const int halfsel = (lane < 16) ? 0 : 1;
  const int nl      = lane & 15;
  const int m0      = blockIdx.x * 256 + wave * 32;
  const int n0      = blockIdx.y * 64;
  const int k0      = halfsel * 8;         // A-fragment K base per ISA layout

  // staging coordinates for this thread (one 16B chunk of the 64x32 tile)
  const int sg_i = threadIdx.x >> 2;       // 0..63 output row in tile
  const int sg_c = (threadIdx.x & 3) * 8;  // 0,8,16,24 k offset

  auto stage = [&](int buf, int kb) {
    const __bf16* wsrc = W + (size_t)(n0 + sg_i) * K + kb + sg_c;
    __bf16* ldst = &lw[buf][sg_i * ROWP + sg_c];
#if HAS_ASYNC_LDS
    __builtin_amdgcn_global_load_async_to_lds_b128(
        (__attribute__((address_space(1))) v4i*)wsrc,
        (__attribute__((address_space(3))) v4i*)ldst, 0, 0);
#else
    v8bf wv = *(const v8bf*)wsrc;
    *(v8bf*)ldst = wv;
#endif
  };

  auto load_afrag = [&](int row, int kb) -> v16bf {
    v16bf a;
    if (A_BF16) {
      const __bf16* arow = (const __bf16*)Av + (size_t)row * K + kb + k0;
      v8bf lo = *(const v8bf*)(arow);
      v8bf hi = *(const v8bf*)(arow + 16);
#pragma unroll
      for (int j = 0; j < 8; ++j) { a[j] = lo[j]; a[j + 8] = hi[j]; }
    } else {
      const float* arow = (const float*)Av + (size_t)row * K + kb + k0;
      v4f f0 = *(const v4f*)(arow);
      v4f f1 = *(const v4f*)(arow + 4);
      v4f f2 = *(const v4f*)(arow + 16);
      v4f f3 = *(const v4f*)(arow + 20);
#pragma unroll
      for (int j = 0; j < 4; ++j) {
        a[j] = (__bf16)f0[j];     a[j + 4]  = (__bf16)f1[j];
        a[j + 8] = (__bf16)f2[j]; a[j + 12] = (__bf16)f3[j];
      }
    }
    return a;
  };

  v8f acc[2][4] = {};
  stage(0, 0);
  int buf = 0;

  for (int kb = 0; kb < K; kb += 32) {
    wait_async_lds();        // my staged writes (prev iter) are in LDS
    __syncthreads();         // everyone's writes visible; prev reads retired
    if (kb + 32 < K) stage(buf ^ 1, kb + 32);

    // A fragments for both M-tiles (+ prefetch next K tile)
    v16bf a0 = load_afrag(m0 + nl, kb);
    v16bf a1 = load_afrag(m0 + 16 + nl, kb);
    if (kb + 32 < K) {
      if (A_BF16)
        __builtin_prefetch((const __bf16*)Av + (size_t)(m0 + nl) * K + kb + 32 + k0, 0, 0);
      else
        __builtin_prefetch((const float*)Av + (size_t)(m0 + nl) * K + kb + 32 + k0, 0, 0);
    }

    // preload ALL B fragments, then issue the 8 WMMAs (batched ds waits)
    v16bf bf[4];
#pragma unroll
    for (int nt = 0; nt < 4; ++nt) {
      const __bf16* brow = &lw[buf][(nt * 16 + nl) * ROWP + halfsel * 8];
      v8bf b0 = *(const v8bf*)(brow);
      v8bf b1 = *(const v8bf*)(brow + 16);
#pragma unroll
      for (int j = 0; j < 8; ++j) { bf[nt][j] = b0[j]; bf[nt][j + 8] = b1[j]; }
    }
#pragma unroll
    for (int nt = 0; nt < 4; ++nt) {
      acc[0][nt] = __builtin_amdgcn_wmma_f32_16x16x32_bf16(
          false, a0, false, bf[nt], (short)0, acc[0][nt], false, false);
      acc[1][nt] = __builtin_amdgcn_wmma_f32_16x16x32_bf16(
          false, a1, false, bf[nt], (short)0, acc[1][nt], false, false);
    }
    buf ^= 1;
  }

  // store D: VGPR r -> row m0 + mt*16 + r + halfsel*8, col n0 + nt*16 + nl
#pragma unroll
  for (int mt = 0; mt < 2; ++mt) {
#pragma unroll
    for (int nt = 0; nt < 4; ++nt) {
      const int col = n0 + nt * 16 + nl;
      const float bv = bias ? bias[col] : 0.f;
#pragma unroll
      for (int r = 0; r < 8; ++r) {
        const int row = m0 + mt * 16 + r + halfsel * 8;
        float v = acc[mt][nt][r] + bv;
        if (RELU) v = fmaxf(v, 0.f);
        if (OUT_BF16)
          ((__bf16*)Cv)[(size_t)row * Nout + col] = (__bf16)v;
        else
          ((float*)Cv)[(size_t)row * Nout + col] = v;
      }
    }
  }
}

// ---------------------------------------------------------------------------
// softmax over 64 experts + top-8 + renorm + balance-loss accumulators
// one wave (32 lanes) per token, 2 experts per lane; 8 tokens per block.
// ---------------------------------------------------------------------------
__global__ __launch_bounds__(256) void softmax_topk_kernel(
    const float* __restrict__ logits, const int* __restrict__ amask,
    float* __restrict__ out_idx_f, float* __restrict__ out_scores,
    int* __restrict__ idx_i, float* __restrict__ g_acc /*cnt[64],ssum[64],nv*/,
    int N) {
  __shared__ float s_cnt[64], s_ssum[64], s_nv[1];
  if (threadIdx.x < 64) { s_cnt[threadIdx.x] = 0.f; s_ssum[threadIdx.x] = 0.f; }
  if (threadIdx.x == 0) s_nv[0] = 0.f;
  __syncthreads();

  const int lane = threadIdx.x & 31;
  const int wave = threadIdx.x >> 5;
  const int tok  = blockIdx.x * 8 + wave;

  if (tok < N) {
    float x0 = logits[(size_t)tok * 64 + lane];
    float x1 = logits[(size_t)tok * 64 + lane + 32];
    float m = fmaxf(x0, x1);
#pragma unroll
    for (int o = 16; o; o >>= 1) m = fmaxf(m, __shfl_xor(m, o, 32));
    float e0 = __expf(x0 - m), e1 = __expf(x1 - m);
    float s = e0 + e1;
#pragma unroll
    for (int o = 16; o; o >>= 1) s += __shfl_xor(s, o, 32);
    const float inv = 1.f / s;
    const float sc0 = e0 * inv, sc1 = e1 * inv;
    const float msk = (float)amask[tok];

    atomicAdd(&s_ssum[lane],      sc0 * msk);
    atomicAdd(&s_ssum[lane + 32], sc1 * msk);
    if (lane == 0) atomicAdd(&s_nv[0], msk);

    // iterative wave argmax for top-8 (ties -> smaller index, like lax.top_k)
    float r0 = sc0, r1 = sc1;
    float tv[NUM_ACTIVE]; int ti[NUM_ACTIVE];
#pragma unroll
    for (int r = 0; r < NUM_ACTIVE; ++r) {
      float v; int idx;
      if (r0 >= r1) { v = r0; idx = lane; } else { v = r1; idx = lane + 32; }
#pragma unroll
      for (int o = 16; o; o >>= 1) {
        float ov = __shfl_xor(v, o, 32);
        int   oi = __shfl_xor(idx, o, 32);
        if (ov > v || (ov == v && oi < idx)) { v = ov; idx = oi; }
      }
      tv[r] = v; ti[r] = idx;
      if (idx == lane)      r0 = -1.f;
      if (idx == lane + 32) r1 = -1.f;
    }
    float tsum = 0.f;
#pragma unroll
    for (int r = 0; r < NUM_ACTIVE; ++r) tsum += tv[r];
    const float tinv = 1.f / tsum;

    if (lane == 0) {
#pragma unroll
      for (int r = 0; r < NUM_ACTIVE; ++r) {
        out_idx_f[(size_t)tok * 8 + r]  = (float)ti[r];
        out_scores[(size_t)tok * 8 + r] = tv[r] * tinv;
        idx_i[(size_t)tok * 8 + r]      = ti[r];
        atomicAdd(&s_cnt[ti[r]], msk);
      }
    }
  }
  __syncthreads();
  if (threadIdx.x < 64) {
    atomicAdd(&g_acc[threadIdx.x],      s_cnt[threadIdx.x]);
    atomicAdd(&g_acc[64 + threadIdx.x], s_ssum[threadIdx.x]);
  }
  if (threadIdx.x == 0) atomicAdd(&g_acc[128], s_nv[0]);
}

// ---------------------------------------------------------------------------
// balance loss + load (1 block, 64 threads)
// ---------------------------------------------------------------------------
__global__ void finalize_kernel(const float* __restrict__ acc,
                                float* __restrict__ loss_out,
                                float* __restrict__ load_out) {
  __shared__ float partial[64];
  const int t = threadIdx.x;
  if (t < 64) {
    const float nv  = acc[128];
    const float cnt = acc[t];
    const float ss  = acc[64 + t];
    partial[t]  = (cnt / nv) * (ss / nv);
    load_out[t] = cnt;            // tokens_per_expert * num_valid == raw count
  }
  __syncthreads();
  if (t == 0) {
    float s = 0.f;
    for (int i = 0; i < 64; ++i) s += partial[i];
    *loss_out = BALANCE_W * (float)NUM_EXPERTS * s;
  }
}

// ---------------------------------------------------------------------------
// selected_keys gather: one block per (token, k) slot, copies 1024 floats.
// Output base is only 4B-aligned (offset 2*N*8+65 floats) -> scalar stores.
// ---------------------------------------------------------------------------
__global__ __launch_bounds__(256) void gather_keys_kernel(
    const float* __restrict__ keys, const int* __restrict__ idx_i,
    float* __restrict__ sel) {
  const int pair = blockIdx.x;                 // token*8 + k
  const int e = idx_i[pair];
  const float* src = keys + (size_t)e * DIM_OUT;
  float* dst = sel + (size_t)pair * DIM_OUT;
  for (int j = threadIdx.x; j < DIM_OUT; j += 256) dst[j] = src[j];
}

// ---------------------------------------------------------------------------
extern "C" void kernel_launch(void* const* d_in, const int* in_sizes, int n_in,
                              void* d_out, int out_size, void* d_ws, size_t ws_size,
                              hipStream_t stream) {
  (void)in_sizes; (void)n_in; (void)out_size; (void)ws_size;

  const float* x    = (const float*)d_in[0];   // [N, 4096]
  const float* w1   = (const float*)d_in[1];   // [1024, 4096]
  const float* b1   = (const float*)d_in[2];   // [1024]
  const float* w2   = (const float*)d_in[3];   // [1024, 1024]
  const float* b2   = (const float*)d_in[4];   // [1024]
  const float* keys = (const float*)d_in[5];   // [64, 1024]
  const int*   am   = (const int*)d_in[6];     // [B,S] -> flat N

  float* out = (float*)d_out;
  const size_t N = NTOK;
  // output layout (floats): idx[N*8] | scores[N*8] | loss[1] | load[64] | sel[N*8*1024]
  float* out_idx    = out;
  float* out_scores = out + N * 8;
  float* out_loss   = out + 2 * N * 8;
  float* out_load   = out + 2 * N * 8 + 1;
  float* out_sel    = out + 2 * N * 8 + 1 + 64;

  // workspace layout
  char* ws = (char*)d_ws;
  __bf16* w1b   = (__bf16*)ws;  ws += (size_t)DIM_HID * DIM_IN * 2;      // 8 MB
  __bf16* w2b   = (__bf16*)ws;  ws += (size_t)DIM_OUT * DIM_HID * 2;     // 2 MB
  __bf16* keysb = (__bf16*)ws;  ws += (size_t)NUM_EXPERTS * DIM_OUT * 2; // 128 KB
  __bf16* h     = (__bf16*)ws;  ws += (size_t)N * DIM_HID * 2;           // 32 MB
  __bf16* g     = (__bf16*)ws;  ws += (size_t)N * DIM_OUT * 2;           // 32 MB
  float*  lg    = (float*)ws;   ws += (size_t)N * NUM_EXPERTS * 4;       // 4 MB
  int*    idxw  = (int*)ws;     ws += (size_t)N * NUM_ACTIVE * 4;        // 512 KB
  float*  acc   = (float*)ws;                                            // cnt64+ssum64+nv

  // 1) convert weights to bf16
  f32_to_bf16_kernel<<<2048, 256, 0, stream>>>(w1, w1b, DIM_HID * DIM_IN);
  f32_to_bf16_kernel<<<1024, 256, 0, stream>>>(w2, w2b, DIM_OUT * DIM_HID);
  f32_to_bf16_kernel<<<64, 256, 0, stream>>>(keys, keysb, NUM_EXPERTS * DIM_OUT);
  zero_accum_kernel<<<1, 256, 0, stream>>>(acc);

  // 2) h = relu(x @ w1^T + b1)   (A fp32 -> bf16 in-register, out bf16)
  gemm_wmma_kernel<false, true, true><<<dim3(N / 256, DIM_HID / 64), 256, 0, stream>>>(
      x, w1b, b1, h, (int)N, DIM_IN, DIM_HID);

  // 3) g = h @ w2^T + b2         (A bf16, out bf16)
  gemm_wmma_kernel<true, true, false><<<dim3(N / 256, DIM_OUT / 64), 256, 0, stream>>>(
      h, w2b, b2, g, (int)N, DIM_HID, DIM_OUT);

  // 4) logits = g @ keys^T       (A bf16, out fp32, Nout = 64)
  gemm_wmma_kernel<true, false, false><<<dim3(N / 256, 1), 256, 0, stream>>>(
      g, keysb, nullptr, lg, (int)N, DIM_OUT, NUM_EXPERTS);

  // 5) softmax + top-8 + accumulators
  softmax_topk_kernel<<<N / 8, 256, 0, stream>>>(lg, am, out_idx, out_scores,
                                                 idxw, acc, (int)N);

  // 6) balance loss + load
  finalize_kernel<<<1, 64, 0, stream>>>(acc, out_loss, out_load);

  // 7) selected_keys gather
  gather_keys_kernel<<<N * NUM_ACTIVE, 256, 0, stream>>>(keys, idxw, out_sel);
}
